// MMTransformer_50354196578496
// MI455X (gfx1250) — matrix operand order repaired and checked by reading, not compiled, not run on hardware
//
#include <hip/hip_runtime.h>

// ---------------- problem constants ----------------
#define BATCH   32
#define IMGL    49
#define TXTL    77
#define SEQL    126           // IMGL + TXTL
#define DMODEL  512
#define NHEAD   8
#define HK      64
#define FFDIM   2048
#define NBLK    4
#define IMGDIM  768
#define VOCAB   49408

#define ROWS_ALL (BATCH * SEQL)   // 4032
#define ROWS_IMG (BATCH * IMGL)   // 1568
#define ROWS_TXT (BATCH * TXTL)   // 2464

#define CDIV(a, b) (((a) + (b) - 1) / (b))
#define NEG_INF (-__builtin_inff())

typedef __attribute__((ext_vector_type(16))) __bf16 v16bf;
typedef __attribute__((ext_vector_type(8)))  __bf16 v8bf;
typedef __attribute__((ext_vector_type(4)))  __bf16 v4bf;
typedef __attribute__((ext_vector_type(2)))  __bf16 v2bf;
typedef __attribute__((ext_vector_type(8)))  float  v8f;
typedef __attribute__((ext_vector_type(2)))  float  vf2;
typedef __attribute__((ext_vector_type(4)))  float  vf4;
typedef __attribute__((ext_vector_type(8)))  float  vf8;

#define CAT16(lo, hi) \
  __builtin_shufflevector(lo, hi, 0, 1, 2, 3, 4, 5, 6, 7, 8, 9, 10, 11, 12, 13, 14, 15)

// native f32 -> bf16 converts (hardware v_cvt_*bf16* on gfx1250)
__device__ __forceinline__ __bf16 f2bf(float f) { return (__bf16)f; }
__device__ __forceinline__ v2bf cvt2(float a, float b) {
  vf2 t = {a, b};
  return __builtin_convertvector(t, v2bf);
}
__device__ __forceinline__ v4bf cvt4(float4 v) {
  vf4 t = {v.x, v.y, v.z, v.w};
  return __builtin_convertvector(t, v4bf);
}
__device__ __forceinline__ v8bf cvt8(float4 a, float4 b) {
  vf8 t = {a.x, a.y, a.z, a.w, b.x, b.y, b.z, b.w};
  return __builtin_convertvector(t, v8bf);
}

__device__ __forceinline__ v8f wmma_bf16(v16bf a, v16bf b, v8f c) {
  // D = A(16x32 bf16) * B(32x16 bf16) + C(16x16 f32)
  return __builtin_amdgcn_wmma_f32_16x16x32_bf16(
      false, a, false, b, (short)0, c, false, false);
}

// A fragment: two contiguous 8-half chunks at p and p+16 (halves), 16B aligned.
__device__ __forceinline__ v16bf afrag(const __bf16* p) {
  return CAT16(*(const v8bf*)p, *(const v8bf*)(p + 16));
}
// B fragment (transposed-B LDS layout): two contiguous 8-half chunks at p and p+16 (halves), 16B aligned.
__device__ __forceinline__ v16bf bfrag(const __bf16* p) {
  return CAT16(*(const v8bf*)p, *(const v8bf*)(p + 16));
}

__device__ __forceinline__ float gelu_exact(float x) {
  return 0.5f * x * (1.0f + erff(x * 0.70710678118654752f));
}

// ---------------- generic bf16-WMMA GEMM, f32 in / f32 out ----------------
// C[M,N] (+)= act(A[M,K] * Bw[K,N] + bias)
// RESID: accumulate into existing C.  ACT: exact GELU.
// BQKV: Bw is [H=8][K][64]; logical column n maps to (n>>6)*K*64 + k*64 + (n&63).
#define BM 128
#define BN 128
#define BK 32

template <int BQKV>
__device__ __forceinline__ void stage_tile(
    __bf16 (*As)[BK + 8], __bf16 (*BsT)[BK + 8],
    const float* __restrict__ A, int lda,
    const float* __restrict__ Bw, int ldb,
    int m0, int n0, int k0, int M, int K, int t) {
  // ---- A tile (128x32): float4 global loads -> v4bf LDS stores ----
#pragma unroll
  for (int i = 0; i < 4; ++i) {
    int idx = t + i * 256;          // 0..1023 units of 4 elems
    int r = idx >> 3;               // 0..127
    int c = (idx & 7) << 2;         // 0..28
    float4 v = {0.f, 0.f, 0.f, 0.f};
    int gr = m0 + r;
    if (gr < M) v = *(const float4*)(A + (size_t)gr * lda + k0 + c);
    *(v4bf*)&As[r][c] = cvt4(v);
  }
  // ---- B tile transposed (BsT[col][k]): paired k-rows -> b32 stores ----
#pragma unroll
  for (int i = 0; i < 2; ++i) {
    int idx = t + i * 256;          // 0..511 units of 2x4 elems
    int r2  = idx >> 5;             // k-row pair id 0..15
    int c   = (idx & 31) << 2;      // 0..124
    int gc  = n0 + c;
    const float *p0, *p1;
    if (BQKV) {
      size_t base = (size_t)(gc >> 6) * ((size_t)K * HK) + (gc & 63);
      p0 = Bw + base + (size_t)(k0 + 2 * r2) * HK;
      p1 = p0 + HK;
    } else {
      p0 = Bw + (size_t)(k0 + 2 * r2) * ldb + gc;
      p1 = p0 + ldb;
    }
    float4 va = *(const float4*)p0;
    float4 vb = *(const float4*)p1;
    *(v2bf*)&BsT[c + 0][2 * r2] = cvt2(va.x, vb.x);
    *(v2bf*)&BsT[c + 1][2 * r2] = cvt2(va.y, vb.y);
    *(v2bf*)&BsT[c + 2][2 * r2] = cvt2(va.z, vb.z);
    *(v2bf*)&BsT[c + 3][2 * r2] = cvt2(va.w, vb.w);
  }
}

template <int RESID, int ACT, int BQKV>
__global__ __launch_bounds__(256) void gemm_ws_kernel(
    const float* __restrict__ A, int lda,
    const float* __restrict__ Bw, int ldb,
    const float* __restrict__ bias,
    float* __restrict__ C, int ldc,
    int M, int N, int K) {
  __shared__ __align__(16) __bf16 As[2][BM][BK + 8];   // double-buffered
  __shared__ __align__(16) __bf16 BsT[2][BN][BK + 8];  // B^T: [col][k]

  const int t    = threadIdx.x;
  const int lane = t & 31;
  const int wave = t >> 5;
  const int wm   = wave >> 2;      // 0..1
  const int wn   = wave & 3;       // 0..3
  const int m0   = blockIdx.y * BM;
  const int n0   = blockIdx.x * BN;
  const int ar   = wm * 64;        // wave's row offset in tile
  const int bc   = wn * 32;        // wave's col offset in tile
  const int colLo = lane & 15;
  const int hi    = (lane >> 4) & 1;

  v8f acc[4][2];
#pragma unroll
  for (int i = 0; i < 4; ++i)
#pragma unroll
    for (int j = 0; j < 2; ++j)
      acc[i][j] = (v8f){0.f, 0.f, 0.f, 0.f, 0.f, 0.f, 0.f, 0.f};

  const int nk = K / BK;
  stage_tile<BQKV>(As[0], BsT[0], A, lda, Bw, ldb, m0, n0, 0, M, K, t);
  __syncthreads();

  for (int kt = 0; kt < nk; ++kt) {
    int cur = kt & 1;
    // pipeline: stage tile kt+1 into the other buffer while computing kt
    if (kt + 1 < nk)
      stage_tile<BQKV>(As[cur ^ 1], BsT[cur ^ 1], A, lda, Bw, ldb, m0, n0,
                       (kt + 1) * BK, M, K, t);
    if (!BQKV && (kt + 2) < nk) {
      // speculative prefetch two tiles ahead -> global_prefetch_b8
      const float* nb =
          Bw + (size_t)((kt + 2) * BK + (t >> 3)) * ldb + n0 + ((t & 7) << 4);
      __builtin_prefetch(nb, 0, 0);
    }

    v16bf bf[2];
#pragma unroll
    for (int sn = 0; sn < 2; ++sn)
      bf[sn] = bfrag(&BsT[cur][bc + sn * 16 + colLo][hi * 8]);
#pragma unroll
    for (int sm = 0; sm < 4; ++sm) {
      v16bf af = afrag(&As[cur][ar + sm * 16 + colLo][hi * 8]);
#pragma unroll
      for (int sn = 0; sn < 2; ++sn)
        acc[sm][sn] = wmma_bf16(af, bf[sn], acc[sm][sn]);
    }
    __syncthreads();  // staging of kt+1 done; safe to overwrite cur next iter
  }

  // epilogue: C-layout is VGPR r -> row r (lanes 0-15) / r+8 (lanes 16-31)
  const int rOff = hi * 8;
#pragma unroll
  for (int sm = 0; sm < 4; ++sm) {
#pragma unroll
    for (int sn = 0; sn < 2; ++sn) {
      int col   = n0 + bc + sn * 16 + colLo;
      int rbase = m0 + ar + sm * 16 + rOff;
      float bv  = bias ? bias[col] : 0.f;
#pragma unroll
      for (int r = 0; r < 8; ++r) {
        int row = rbase + r;
        if (row < M) {
          float v = acc[sm][sn][r] + bv;
          if (ACT) v = gelu_exact(v);
          size_t o = (size_t)row * ldc + col;
          if (RESID) v += C[o];
          C[o] = v;
        }
      }
    }
  }
}

// ---------------- fused attention (one block per (b,h)) ----------------
__global__ __launch_bounds__(256) void attn_kernel(
    const float* __restrict__ Q, const float* __restrict__ Kb,
    const float* __restrict__ V, const int* __restrict__ amask,
    float* __restrict__ O) {
  // U holds Ksn[128][72] (K rows, natural layout == transposed-B for Q*K^T)
  // during scores, then is reused as per-wave probability tiles AW[8][16][136].
  __shared__ __align__(16) __bf16 U[17408];
  __shared__ __align__(16) __bf16 Vt[64][136];  // V^T: [hk][n]
  __shared__ float mlds[128];

  const int bh   = blockIdx.x;
  const int b    = bh >> 3;
  const int hh   = bh & 7;
  const int t    = threadIdx.x;
  const int lane = t & 31;
  const int wave = t >> 5;
  const size_t headBase = ((size_t)b * SEQL) * DMODEL + hh * HK;

  // stage K rows (vectorized direct copy; pad rows >= SEQL with zero)
#pragma unroll
  for (int i = 0; i < 8; ++i) {
    int idx = t + i * 256;           // 0..2047 units of 4
    int n  = idx >> 4;               // 0..127
    int kb = (idx & 15) << 2;        // 0..60
    float4 v = {0.f, 0.f, 0.f, 0.f};
    if (n < SEQL) v = *(const float4*)(Kb + headBase + (size_t)n * DMODEL + kb);
    *(v4bf*)&U[n * 72 + kb] = cvt4(v);
  }
  // stage V transposed (paired n-rows -> b32 stores)
#pragma unroll
  for (int i = 0; i < 4; ++i) {
    int idx = t + i * 256;           // 0..1023 units of 2x4
    int m  = idx >> 4;               // n pair id 0..63
    int kb = (idx & 15) << 2;        // 0..60
    int n2 = 2 * m;
    float4 va = {0.f, 0.f, 0.f, 0.f}, vb = {0.f, 0.f, 0.f, 0.f};
    if (n2 < SEQL)     va = *(const float4*)(V + headBase + (size_t)n2 * DMODEL + kb);
    if (n2 + 1 < SEQL) vb = *(const float4*)(V + headBase + (size_t)(n2 + 1) * DMODEL + kb);
    *(v2bf*)&Vt[kb + 0][n2] = cvt2(va.x, vb.x);
    *(v2bf*)&Vt[kb + 1][n2] = cvt2(va.y, vb.y);
    *(v2bf*)&Vt[kb + 2][n2] = cvt2(va.z, vb.z);
    *(v2bf*)&Vt[kb + 3][n2] = cvt2(va.w, vb.w);
  }
  if (t < 128) {
    float madd = 0.f;
    if (t >= SEQL) madd = NEG_INF;
    else if (t >= IMGL && amask[b * TXTL + (t - IMGL)] == 0) madd = NEG_INF;
    mlds[t] = madd;
  }
  __syncthreads();

  // per-wave: 16 query rows
  const int qbase = wave * 16;
  const int colLo = lane & 15;
  const int hi    = (lane >> 4) & 1;
  const int rOff  = hi * 8;

  // Q fragments straight from global (f32 float4 loads -> bf16), 2 k-steps
  int qr  = qbase + colLo;
  int qrc = qr < SEQL ? qr : SEQL - 1;
  const float* qp = Q + headBase + (size_t)qrc * DMODEL;
  v16bf aq[2];
#pragma unroll
  for (int ks = 0; ks < 2; ++ks) {
    const float* qk = qp + ks * 32 + hi * 8;
    float4 a0 = *(const float4*)(qk);
    float4 a1 = *(const float4*)(qk + 4);
    float4 b0 = *(const float4*)(qk + 16);
    float4 b1 = *(const float4*)(qk + 20);
    aq[ks] = CAT16(cvt8(a0, a1), cvt8(b0, b1));
  }

  // scores: 8 col-tiles x 2 k-steps; B fragments contiguous from Ksn
  v8f sc[8];
#pragma unroll
  for (int tn = 0; tn < 8; ++tn) {
    sc[tn] = (v8f){0.f, 0.f, 0.f, 0.f, 0.f, 0.f, 0.f, 0.f};
#pragma unroll
    for (int ks = 0; ks < 2; ++ks)
      sc[tn] = wmma_bf16(aq[ks],
                         bfrag(&U[(tn * 16 + colLo) * 72 + ks * 32 + hi * 8]),
                         sc[tn]);
  }
  __syncthreads();  // everyone done reading Ksn before U is reused for probs

  // softmax per row (16 lanes share a row; xor masks stay inside the group)
  const float scale = 0.044194173824159216f;  // D**-0.5 (faithful to ref)
  __bf16* AW = &U[(wave * 16) * 136];
#pragma unroll
  for (int r = 0; r < 8; ++r) {
    int qrow = qbase + rOff + r;
    float vals[8];
    float mx = NEG_INF;
#pragma unroll
    for (int tn = 0; tn < 8; ++tn) {
      int col = tn * 16 + colLo;
      float s = sc[tn][r] * scale + mlds[col];
      if (qrow >= IMGL && col >= IMGL && col > qrow) s = NEG_INF;  // causal txt
      vals[tn] = s;
      mx = fmaxf(mx, s);
    }
#pragma unroll
    for (int m = 1; m < 16; m <<= 1) mx = fmaxf(mx, __shfl_xor(mx, m));
    float sum = 0.f;
#pragma unroll
    for (int tn = 0; tn < 8; ++tn) {
      float e = __expf(vals[tn] - mx);
      vals[tn] = e;
      sum += e;
    }
#pragma unroll
    for (int m = 1; m < 16; m <<= 1) sum += __shfl_xor(sum, m);
    float inv = 1.0f / sum;
#pragma unroll
    for (int tn = 0; tn < 8; ++tn)
      AW[(rOff + r) * 136 + tn * 16 + colLo] = f2bf(vals[tn] * inv);
  }

  // O = P (16x128) * V (128x64): 4 k-steps x 4 col-tiles, all b128 frags
  v8f oacc[4];
#pragma unroll
  for (int sn = 0; sn < 4; ++sn)
    oacc[sn] = (v8f){0.f, 0.f, 0.f, 0.f, 0.f, 0.f, 0.f, 0.f};
#pragma unroll
  for (int ks = 0; ks < 4; ++ks) {
    v16bf ap = afrag(AW + colLo * 136 + ks * 32 + hi * 8);
#pragma unroll
    for (int sn = 0; sn < 4; ++sn)
      oacc[sn] = wmma_bf16(ap, bfrag(&Vt[sn * 16 + colLo][ks * 32 + hi * 8]),
                           oacc[sn]);
  }
#pragma unroll
  for (int sn = 0; sn < 4; ++sn) {
    int col = sn * 16 + colLo;
#pragma unroll
    for (int r = 0; r < 8; ++r) {
      int row = qbase + rOff + r;
      if (row < SEQL) O[headBase + (size_t)row * DMODEL + col] = oacc[sn][r];
    }
  }
}

// ---------------- LayerNorm: one wave per row (D=512 -> 16/lane) ----------
__global__ __launch_bounds__(256) void ln_kernel(
    const float* __restrict__ X, const float* __restrict__ g,
    const float* __restrict__ bb, float* __restrict__ Y, int rows,
    int txtOnly) {
  int row = blockIdx.x * 8 + (threadIdx.x >> 5);
  if (row >= rows) return;
  int lane  = threadIdx.x & 31;
  int inRow = row;
  if (txtOnly) {
    int bt = row / TXTL, tt = row - bt * TXTL;
    inRow = bt * SEQL + IMGL + tt;
  }
  const float* xp = X + (size_t)inRow * DMODEL;
  float vals[16];
  float s = 0.f;
#pragma unroll
  for (int i = 0; i < 16; ++i) {
    float v = xp[lane + 32 * i];
    vals[i] = v;
    s += v;
  }
#pragma unroll
  for (int m = 1; m < 32; m <<= 1) s += __shfl_xor(s, m);
  float mean = s * (1.0f / DMODEL);
  float s2 = 0.f;
#pragma unroll
  for (int i = 0; i < 16; ++i) {
    float d = vals[i] - mean;
    s2 += d * d;
  }
#pragma unroll
  for (int m = 1; m < 32; m <<= 1) s2 += __shfl_xor(s2, m);
  float rstd = rsqrtf(s2 * (1.0f / DMODEL) + 1e-5f);
  float* yp = Y + (size_t)row * DMODEL;
#pragma unroll
  for (int i = 0; i < 16; ++i) {
    int j = lane + 32 * i;
    yp[j] = (vals[i] - mean) * rstd * g[j] + bb[j];
  }
}

// ---------------- embedding assembly ----------------
__global__ void scatter_img_kernel(const float* __restrict__ ximg,
                                   const float* __restrict__ pos,
                                   float* __restrict__ x) {
  int idx = blockIdx.x * blockDim.x + threadIdx.x;
  int total = ROWS_IMG * DMODEL;
  if (idx >= total) return;
  int m = idx >> 9, d = idx & 511;
  int b = m / IMGL, l = m - b * IMGL;
  x[((size_t)b * SEQL + l) * DMODEL + d] = ximg[idx] + pos[l * DMODEL + d];
}

__global__ void embed_txt_kernel(const float* __restrict__ txt,
                                 const float* __restrict__ pos,
                                 float* __restrict__ x) {
  int idx = blockIdx.x * blockDim.x + threadIdx.x;
  int total = ROWS_TXT * DMODEL;
  if (idx >= total) return;
  int m = idx >> 9, d = idx & 511;
  int b = m / TXTL, tt = m - b * TXTL;
  x[((size_t)b * SEQL + IMGL + tt) * DMODEL + d] =
      txt[idx] + pos[(IMGL + tt) * DMODEL + d];
}

// ---------------- driver ----------------
extern "C" void kernel_launch(void* const* d_in, const int* in_sizes, int n_in,
                              void* d_out, int out_size, void* d_ws,
                              size_t ws_size, hipStream_t stream) {
  const float* img_hidden = (const float*)d_in[0];
  const float* txt_embed  = (const float*)d_in[1];
  const float* img_proj_w = (const float*)d_in[2];
  const float* img_proj_b = (const float*)d_in[3];
  const float* pos_embed  = (const float*)d_in[4];
  const float* wq     = (const float*)d_in[5];
  const float* wk     = (const float*)d_in[6];
  const float* wv     = (const float*)d_in[7];
  const float* w_proj = (const float*)d_in[8];
  const float* b_proj = (const float*)d_in[9];
  const float* ln1_g = (const float*)d_in[10];
  const float* ln1_b = (const float*)d_in[11];
  const float* ln2_g = (const float*)d_in[12];
  const float* ln2_b = (const float*)d_in[13];
  const float* fc1_w = (const float*)d_in[14];
  const float* fc1_b = (const float*)d_in[15];
  const float* fc2_w = (const float*)d_in[16];
  const float* fc2_b = (const float*)d_in[17];
  const float* lnf_g = (const float*)d_in[18];
  const float* lnf_b = (const float*)d_in[19];
  const float* out_w = (const float*)d_in[20];
  const float* out_b = (const float*)d_in[21];
  const int*   amask = (const int*)d_in[22];
  (void)in_sizes; (void)n_in; (void)out_size; (void)ws_size;

  float* ws = (float*)d_ws;
  const size_t SZ_X = (size_t)ROWS_ALL * DMODEL;
  float* x    = ws;                 // residual stream [4032,512]
  float* h    = x + SZ_X;           // LN out / attention out (reused)
  float* q    = h + SZ_X;
  float* k    = q + SZ_X;
  float* v    = k + SZ_X;
  float* a1   = v + SZ_X;           // [4032,2048]
  float* ximg = a1 + (size_t)ROWS_ALL * FFDIM;     // [1568,512]
  float* hf   = ximg + (size_t)ROWS_IMG * DMODEL;  // [2464,512]

  dim3 blk(256);

  // ---- embedding ----
  gemm_ws_kernel<0, 0, 0><<<dim3(DMODEL / BN, CDIV(ROWS_IMG, BM)), blk, 0,
                            stream>>>(img_hidden, IMGDIM, img_proj_w, DMODEL,
                                      img_proj_b, ximg, DMODEL, ROWS_IMG,
                                      DMODEL, IMGDIM);
  scatter_img_kernel<<<CDIV(ROWS_IMG * DMODEL, 256), blk, 0, stream>>>(
      ximg, pos_embed, x);
  embed_txt_kernel<<<CDIV(ROWS_TXT * DMODEL, 256), blk, 0, stream>>>(
      txt_embed, pos_embed, x);

  // ---- transformer blocks ----
  for (int i = 0; i < NBLK; ++i) {
    const size_t wOff = (size_t)i * NHEAD * DMODEL * HK;  // wq/wk/wv per block
    ln_kernel<<<CDIV(ROWS_ALL, 8), blk, 0, stream>>>(
        x, ln1_g + i * DMODEL, ln1_b + i * DMODEL, h, ROWS_ALL, 0);
    gemm_ws_kernel<0, 0, 1><<<dim3(DMODEL / BN, CDIV(ROWS_ALL, BM)), blk, 0,
                              stream>>>(h, DMODEL, wq + wOff, 0, nullptr, q,
                                        DMODEL, ROWS_ALL, DMODEL, DMODEL);
    gemm_ws_kernel<0, 0, 1><<<dim3(DMODEL / BN, CDIV(ROWS_ALL, BM)), blk, 0,
                              stream>>>(h, DMODEL, wk + wOff, 0, nullptr, k,
                                        DMODEL, ROWS_ALL, DMODEL, DMODEL);
    gemm_ws_kernel<0, 0, 1><<<dim3(DMODEL / BN, CDIV(ROWS_ALL, BM)), blk, 0,
                              stream>>>(h, DMODEL, wv + wOff, 0, nullptr, v,
                                        DMODEL, ROWS_ALL, DMODEL, DMODEL);
    attn_kernel<<<BATCH * NHEAD, blk, 0, stream>>>(q, k, v, amask, h);
    gemm_ws_kernel<1, 0, 0><<<dim3(DMODEL / BN, CDIV(ROWS_ALL, BM)), blk, 0,
                              stream>>>(h, DMODEL,
                                        w_proj + (size_t)i * DMODEL * DMODEL,
                                        DMODEL, b_proj + i * DMODEL, x, DMODEL,
                                        ROWS_ALL, DMODEL, DMODEL);
    ln_kernel<<<CDIV(ROWS_ALL, 8), blk, 0, stream>>>(
        x, ln2_g + i * DMODEL, ln2_b + i * DMODEL, h, ROWS_ALL, 0);
    gemm_ws_kernel<0, 1, 0><<<dim3(FFDIM / BN, CDIV(ROWS_ALL, BM)), blk, 0,
                              stream>>>(h, DMODEL,
                                        fc1_w + (size_t)i * DMODEL * FFDIM,
                                        FFDIM, fc1_b + i * FFDIM, a1, FFDIM,
                                        ROWS_ALL, FFDIM, DMODEL);
    gemm_ws_kernel<1, 0, 0><<<dim3(DMODEL / BN, CDIV(ROWS_ALL, BM)), blk, 0,
                              stream>>>(a1, FFDIM,
                                        fc2_w + (size_t)i * FFDIM * DMODEL,
                                        DMODEL, fc2_b + i * DMODEL, x, DMODEL,
                                        ROWS_ALL, DMODEL, FFDIM);
  }

  // ---- final LN (text rows only, compacted) + vocab projection ----
  ln_kernel<<<CDIV(ROWS_TXT, 8), blk, 0, stream>>>(x, lnf_g, lnf_b, hf,
                                                   ROWS_TXT, 1);
  gemm_ws_kernel<0, 0, 0><<<dim3(VOCAB / BN, CDIV(ROWS_TXT, BM)), blk, 0,
                            stream>>>(hf, DMODEL, out_w, VOCAB, out_b,
                                      (float*)d_out, VOCAB, ROWS_TXT, VOCAB,
                                      DMODEL);
}
